// SELFATT_Block_72078141161548
// MI455X (gfx1250) — compile-verified, matrix operand order checked by reading, not run on hardware
//
#include <hip/hip_runtime.h>
#include <hip/hip_bf16.h>

typedef _Float16 v16h __attribute__((ext_vector_type(16)));
typedef _Float16 v8h  __attribute__((ext_vector_type(8)));
typedef float    v8f  __attribute__((ext_vector_type(8)));

#define NHEADS 8
#define HD 32
#define CDIM 256
#define NTOK 64

// LDS layout (bytes); strides chosen so row-to-row offset == 4 DWORDs mod 64 banks (36 for STR_VT/STR_P)
#define STR_XN 264   // halves, 64 rows  (normalized X, later attention output)
#define STR_QK 520   // halves, 64 rows  (Q cols 0..255, K cols 256..511)
#define STR_VT 72    // halves, 256 rows (V transposed: channel-major)
#define STR_P  72    // halves, 16 rows per wave
#define STR_Y  260   // floats, 64 rows  (proj output, overlaps QK region)
#define OFF_QK 33792
#define OFF_VT (33792 + 66560)
#define OFF_P  (33792 + 66560 + 36864)
#define SMEM_BYTES (33792 + 66560 + 36864 + 18432)   // 155648 B -> 2 blocks/WGP

__device__ __forceinline__ v16h load_frag(const _Float16* tile, int stride, int lane) {
  // 16-bit A/B operand layout: lane l holds row (l&15); halves 0..7 = K chunk at
  // k0 = (l>=16 ? 8 : 0), halves 8..15 = K chunk at k0+16.
  const int row = lane & 15;
  const int k0  = (lane >> 4) << 3;
  const _Float16* p = tile + row * stride + k0;
  v8h lo = *(const v8h*)(p);
  v8h hi = *(const v8h*)(p + 16);
  v16h f;
#pragma unroll
  for (int i = 0; i < 8; ++i) { f[i] = lo[i]; f[8 + i] = hi[i]; }
  return f;
}

__device__ __forceinline__ v8f wmma_f16(v16h a, v16h b, v8f c) {
  return __builtin_amdgcn_wmma_f32_16x16x32_f16(false, a, false, b, (short)0, c, false, false);
}

__device__ __forceinline__ float wave_red_add(float v) {
#pragma unroll
  for (int m = 16; m >= 1; m >>= 1) v += __shfl_xor(v, m, 32);
  return v;
}
__device__ __forceinline__ float half_red_max(float v) {
#pragma unroll
  for (int m = 1; m < 16; m <<= 1) v = fmaxf(v, __shfl_xor(v, m, 32));
  return v;
}
__device__ __forceinline__ float half_red_add(float v) {
#pragma unroll
  for (int m = 1; m < 16; m <<= 1) v += __shfl_xor(v, m, 32);
  return v;
}

// Prep: f32 -> f16 with transpose so GEMM B-operands load with the A-pattern.
__global__ void prep_weights(const float* __restrict__ qkv_w,   // (256, 768)
                             const float* __restrict__ proj_w,  // (256, 256)
                             _Float16* __restrict__ qkvT,       // (768, 256)
                             _Float16* __restrict__ projT) {    // (256, 256)
  int i = blockIdx.x * 256 + threadIdx.x;
  if (i < 768 * 256) {
    int n = i >> 8, k = i & 255;
    qkvT[i] = (_Float16)qkv_w[k * 768 + n];
  }
  int j = i - 768 * 256;
  if (j >= 0 && j < 256 * 256) {
    int n = j >> 8, k = j & 255;
    projT[j] = (_Float16)proj_w[k * 256 + n];
  }
}

__global__ __launch_bounds__(256)
void swin_block_kernel(const float* __restrict__ x,
                       const float* __restrict__ n1g, const float* __restrict__ n1b,
                       const _Float16* __restrict__ qkvT, const _Float16* __restrict__ projT,
                       const float* __restrict__ proj_b,
                       const float* __restrict__ n2g, const float* __restrict__ n2b,
                       const float* __restrict__ fc1_w, const float* __restrict__ fc1_b,
                       const float* __restrict__ fc2_w, const float* __restrict__ fc2_b,
                       float* __restrict__ out) {
  extern __shared__ char smem[];
  _Float16* lds_xn = (_Float16*)smem;              // 64 x STR_XN halves
  _Float16* lds_qk = (_Float16*)(smem + OFF_QK);   // 64 x STR_QK halves
  _Float16* lds_vT = (_Float16*)(smem + OFF_VT);   // 256 x STR_VT halves
  _Float16* lds_p  = (_Float16*)(smem + OFF_P);    // 8 waves x 16 x STR_P halves
  float*    lds_y  = (float*)(smem + OFF_QK);      // 64 x STR_Y floats (reuse)

  const int win  = blockIdx.x;           // 0..4095, order (b, hb, wb)
  const int b    = win >> 10;
  const int hb   = (win >> 5) & 31;
  const int wb   = win & 31;
  const int lane = threadIdx.x & 31;
  const int w    = threadIdx.x >> 5;     // wave 0..7
  const int nl   = lane & 15;
  const int mh   = (lane >> 4) << 3;     // C-layout row offset for upper half-wave

  // ---------------- Phase 1: LayerNorm1 -> f16 X in LDS ----------------
  {
    const int cb = lane * 8;
    float g1v[8], b1v[8];
#pragma unroll
    for (int j = 0; j < 8; ++j) { g1v[j] = n1g[cb + j]; b1v[j] = n1b[cb + j]; }
#pragma unroll 1
    for (int i = 0; i < 8; ++i) {
      const int t = w * 8 + i;
      const int r = t >> 3, c = t & 7;
      const size_t row = (size_t)b * 65536 + (size_t)(hb * 8 + r) * 256 + (wb * 8 + c);
      const float* xr = x + row * 256;
      float4 a0 = ((const float4*)xr)[lane * 2];
      float4 a1 = ((const float4*)xr)[lane * 2 + 1];
      float v[8] = {a0.x, a0.y, a0.z, a0.w, a1.x, a1.y, a1.z, a1.w};
      float s = 0.f, ss = 0.f;
#pragma unroll
      for (int j = 0; j < 8; ++j) { s += v[j]; ss += v[j] * v[j]; }
      s  = wave_red_add(s);
      ss = wave_red_add(ss);
      const float mean = s * (1.f / 256.f);
      const float var  = ss * (1.f / 256.f) - mean * mean;
      const float rstd = rsqrtf(var + 1e-5f);
      v8h hv;
#pragma unroll
      for (int j = 0; j < 8; ++j)
        hv[j] = (_Float16)((v[j] - mean) * rstd * g1v[j] + b1v[j]);
      *(v8h*)(lds_xn + t * STR_XN + cb) = hv;
    }
  }
  __syncthreads();

  // ---------------- Phase 2: QKV GEMM (64x256) x (256x768) ----------------
  // Wave owns N-tile pairs {w,w+8},{w+16,w+24},{w+32,w+40} -> pairs are wholly
  // inside the Q / K / V sections. One pass over the weights per block; each
  // k-step issues 8 independent WMMAs from 4 hoisted A frags + 2 B frags.
#pragma unroll 1
  for (int np = 0; np < 3; ++np) {
    const int nt0 = w + np * 16;        // pair: nt0, nt0+8
    const _Float16* Bb0 = qkvT + (size_t)nt0 * 16 * 256;
    const _Float16* Bb1 = Bb0 + 8 * 16 * 256;
    if (np < 2) __builtin_prefetch(qkvT + (size_t)(nt0 + 16) * 16 * 256, 0, 1);
    v8f acc0[4] = {}, acc1[4] = {};
#pragma unroll
    for (int kk = 0; kk < 8; ++kk) {
      v16h b0 = load_frag(Bb0 + kk * 32, 256, lane);
      v16h b1 = load_frag(Bb1 + kk * 32, 256, lane);
      v16h a0 = load_frag(lds_xn + 0 * 16 * STR_XN + kk * 32, STR_XN, lane);
      v16h a1 = load_frag(lds_xn + 1 * 16 * STR_XN + kk * 32, STR_XN, lane);
      v16h a2 = load_frag(lds_xn + 2 * 16 * STR_XN + kk * 32, STR_XN, lane);
      v16h a3 = load_frag(lds_xn + 3 * 16 * STR_XN + kk * 32, STR_XN, lane);
      acc0[0] = wmma_f16(a0, b0, acc0[0]);
      acc0[1] = wmma_f16(a1, b0, acc0[1]);
      acc0[2] = wmma_f16(a2, b0, acc0[2]);
      acc0[3] = wmma_f16(a3, b0, acc0[3]);
      acc1[0] = wmma_f16(a0, b1, acc1[0]);
      acc1[1] = wmma_f16(a1, b1, acc1[1]);
      acc1[2] = wmma_f16(a2, b1, acc1[2]);
      acc1[3] = wmma_f16(a3, b1, acc1[3]);
    }
    if (np < 2) {
      // Q section (np==0): cols nt*16 ; K section (np==1): cols 256+(nt-16)*16
      const int coff0 = (np == 0) ? nt0 * 16 : (256 + (nt0 - 16) * 16);
#pragma unroll
      for (int mt = 0; mt < 4; ++mt)
#pragma unroll
        for (int r = 0; r < 8; ++r) {
          lds_qk[(mt * 16 + mh + r) * STR_QK + coff0 + nl]       = (_Float16)acc0[mt][r];
          lds_qk[(mt * 16 + mh + r) * STR_QK + coff0 + 128 + nl] = (_Float16)acc1[mt][r];
        }
    } else {
      const int ch0 = (nt0 - 32) * 16 + nl;  // V channel
#pragma unroll
      for (int mt = 0; mt < 4; ++mt)
#pragma unroll
        for (int r = 0; r < 8; ++r) {
          lds_vT[ch0 * STR_VT + mt * 16 + mh + r]         = (_Float16)acc0[mt][r];
          lds_vT[(ch0 + 128) * STR_VT + mt * 16 + mh + r] = (_Float16)acc1[mt][r];
        }
    }
  }
  __syncthreads();

  // ---------------- Phase 3: attention, wave w owns head w ----------------
  {
    const float scale = 0.1767766952966369f;  // 1/sqrt(32)
    _Float16* Pw = lds_p + w * 16 * STR_P;
    // K and V^T fragments are invariant across M-tiles: hoist into registers.
    v16h kf[4];
#pragma unroll
    for (int nt = 0; nt < 4; ++nt)
      kf[nt] = load_frag(lds_qk + (nt * 16) * STR_QK + 256 + w * HD, STR_QK, lane);
    v16h vb[4];  // [nt2*2 + ks]
#pragma unroll
    for (int i = 0; i < 4; ++i)
      vb[i] = load_frag(lds_vT + (w * HD + (i >> 1) * 16) * STR_VT + (i & 1) * 32,
                        STR_VT, lane);
#pragma unroll 1
    for (int mt = 0; mt < 4; ++mt) {
      v16h qf = load_frag(lds_qk + (mt * 16) * STR_QK + w * HD, STR_QK, lane);
      v8f S[4];
#pragma unroll
      for (int nt = 0; nt < 4; ++nt) {
        v8f z = {};
        S[nt] = wmma_f16(qf, kf[nt], z);
      }
#pragma unroll
      for (int nt = 0; nt < 4; ++nt)
#pragma unroll
        for (int r = 0; r < 8; ++r) S[nt][r] *= scale;
      // row-wise softmax: lane = column, VGPR r = row (+8 for upper half-wave)
#pragma unroll
      for (int r = 0; r < 8; ++r) {
        float mx = fmaxf(fmaxf(S[0][r], S[1][r]), fmaxf(S[2][r], S[3][r]));
        mx = half_red_max(mx);
        float sum = 0.f;
#pragma unroll
        for (int nt = 0; nt < 4; ++nt) {
          float e = __expf(S[nt][r] - mx);
          S[nt][r] = e; sum += e;
        }
        sum = half_red_add(sum);
        const float inv = 1.0f / sum;
#pragma unroll
        for (int nt = 0; nt < 4; ++nt) S[nt][r] *= inv;
      }
      // P (16x64 f16) to per-wave LDS scratch, then back as A operand
#pragma unroll
      for (int nt = 0; nt < 4; ++nt)
#pragma unroll
        for (int r = 0; r < 8; ++r)
          Pw[(mh + r) * STR_P + nt * 16 + nl] = (_Float16)S[nt][r];
      v16h pa0 = load_frag(Pw + 0 * 32, STR_P, lane);
      v16h pa1 = load_frag(Pw + 1 * 32, STR_P, lane);
#pragma unroll
      for (int nt2 = 0; nt2 < 2; ++nt2) {
        v8f o = {};
        o = wmma_f16(pa0, vb[nt2 * 2 + 0], o);
        o = wmma_f16(pa1, vb[nt2 * 2 + 1], o);
#pragma unroll
        for (int r = 0; r < 8; ++r)
          lds_xn[(mt * 16 + mh + r) * STR_XN + w * HD + nt2 * 16 + nl] = (_Float16)o[r];
      }
    }
  }
  __syncthreads();

  // ---------------- Phase 4a: proj GEMM (+bias) -> f32 Y in LDS ----------------
  // Single pass: wave owns N-tile pair {w, w+8}, 8 accumulators.
  {
    const _Float16* Bb0 = projT + (size_t)w * 16 * 256;
    const _Float16* Bb1 = Bb0 + 8 * 16 * 256;
    v8f acc0[4] = {}, acc1[4] = {};
#pragma unroll
    for (int kk = 0; kk < 8; ++kk) {
      v16h b0 = load_frag(Bb0 + kk * 32, 256, lane);
      v16h b1 = load_frag(Bb1 + kk * 32, 256, lane);
      v16h a0 = load_frag(lds_xn + 0 * 16 * STR_XN + kk * 32, STR_XN, lane);
      v16h a1 = load_frag(lds_xn + 1 * 16 * STR_XN + kk * 32, STR_XN, lane);
      v16h a2 = load_frag(lds_xn + 2 * 16 * STR_XN + kk * 32, STR_XN, lane);
      v16h a3 = load_frag(lds_xn + 3 * 16 * STR_XN + kk * 32, STR_XN, lane);
      acc0[0] = wmma_f16(a0, b0, acc0[0]);
      acc0[1] = wmma_f16(a1, b0, acc0[1]);
      acc0[2] = wmma_f16(a2, b0, acc0[2]);
      acc0[3] = wmma_f16(a3, b0, acc0[3]);
      acc1[0] = wmma_f16(a0, b1, acc1[0]);
      acc1[1] = wmma_f16(a1, b1, acc1[1]);
      acc1[2] = wmma_f16(a2, b1, acc1[2]);
      acc1[3] = wmma_f16(a3, b1, acc1[3]);
    }
    const float pb0 = proj_b[w * 16 + nl];
    const float pb1 = proj_b[(w + 8) * 16 + nl];
#pragma unroll
    for (int mt = 0; mt < 4; ++mt)
#pragma unroll
      for (int r = 0; r < 8; ++r) {
        lds_y[(mt * 16 + mh + r) * STR_Y + w * 16 + nl]        = acc0[mt][r] + pb0;
        lds_y[(mt * 16 + mh + r) * STR_Y + (w + 8) * 16 + nl]  = acc1[mt][r] + pb1;
      }
  }
  __syncthreads();

  // ---------------- Phase 4b: LN2 + MLP(hidden=1, exact GELU) + store ----------------
  {
    const int cb = lane * 8;
    float g2v[8], b2v[8], f1[8], f2[8], fb2[8];
#pragma unroll
    for (int j = 0; j < 8; ++j) {
      g2v[j] = n2g[cb + j]; b2v[j] = n2b[cb + j];
      f1[j] = fc1_w[cb + j]; f2[j] = fc2_w[cb + j]; fb2[j] = fc2_b[cb + j];
    }
    const float fc1b = fc1_b[0];
#pragma unroll 1
    for (int i = 0; i < 8; ++i) {
      const int t = w * 8 + i;
      const float* yr = lds_y + t * STR_Y;
      float4 y0 = ((const float4*)yr)[lane * 2];
      float4 y1 = ((const float4*)yr)[lane * 2 + 1];
      float v[8] = {y0.x, y0.y, y0.z, y0.w, y1.x, y1.y, y1.z, y1.w};
      float s = 0.f, ss = 0.f;
#pragma unroll
      for (int j = 0; j < 8; ++j) { s += v[j]; ss += v[j] * v[j]; }
      s  = wave_red_add(s);
      ss = wave_red_add(ss);
      const float mean = s * (1.f / 256.f);
      const float var  = ss * (1.f / 256.f) - mean * mean;
      const float rstd = rsqrtf(var + 1e-5f);
      float dot = 0.f;
#pragma unroll
      for (int j = 0; j < 8; ++j) {
        float h = (v[j] - mean) * rstd * g2v[j] + b2v[j];
        dot += h * f1[j];
      }
      dot = wave_red_add(dot);
      const float zz = dot + fc1b;
      const float gl = 0.5f * zz * (1.0f + erff(zz * 0.70710678118654752f));
      float4 o0, o1;
      o0.x = gl * f2[0] + fb2[0]; o0.y = gl * f2[1] + fb2[1];
      o0.z = gl * f2[2] + fb2[2]; o0.w = gl * f2[3] + fb2[3];
      o1.x = gl * f2[4] + fb2[4]; o1.y = gl * f2[5] + fb2[5];
      o1.z = gl * f2[6] + fb2[6]; o1.w = gl * f2[7] + fb2[7];
      float* orow = out + ((size_t)win * 64 + t) * 256 + cb;
      ((float4*)orow)[0] = o0;
      ((float4*)orow)[1] = o1;
    }
  }
}

extern "C" void kernel_launch(void* const* d_in, const int* in_sizes, int n_in,
                              void* d_out, int out_size, void* d_ws, size_t ws_size,
                              hipStream_t stream) {
  const float* x      = (const float*)d_in[0];
  // d_in[1], d_in[2] = H, W (256, 256) — shapes hardcoded
  const float* n1g    = (const float*)d_in[3];
  const float* n1b    = (const float*)d_in[4];
  const float* qkv_w  = (const float*)d_in[5];
  const float* proj_w = (const float*)d_in[6];
  const float* proj_b = (const float*)d_in[7];
  const float* n2g    = (const float*)d_in[8];
  const float* n2b    = (const float*)d_in[9];
  const float* fc1_w  = (const float*)d_in[10];
  const float* fc1_b  = (const float*)d_in[11];
  const float* fc2_w  = (const float*)d_in[12];
  const float* fc2_b  = (const float*)d_in[13];

  _Float16* qkvT  = (_Float16*)d_ws;            // 768*256 halves
  _Float16* projT = qkvT + 768 * 256;           // 256*256 halves

  prep_weights<<<1024, 256, 0, stream>>>(qkv_w, proj_w, qkvT, projT);

  swin_block_kernel<<<4096, 256, SMEM_BYTES, stream>>>(
      x, n1g, n1b, qkvT, projT, proj_b, n2g, n2b,
      fc1_w, fc1_b, fc2_w, fc2_b, (float*)d_out);
}
